// SelfAttention_68573447848208
// MI455X (gfx1250) — matrix operand order repaired and checked by reading, not compiled, not run on hardware
//
#include <hip/hip_runtime.h>
#include <hip/hip_fp16.h>

// ---------------------------------------------------------------------------
// Causal self-attention for MI455X (gfx1250, wave32, WMMA f16 16x16x32).
//
// Prep:   x -> f16 Xh;  Wq/Wk/Wv -> f16 transposed Wt[n][d] (LDS tile transpose)
// GEMMs:  all reduced to one pattern  C = A * B^T  with row-major f16 A and B:
//           Q[m][n]    = Xh[m]   . Wt_q[n]
//           K[m][n]    = Xh[m]   . Wt_k[n]
//           Vt[v][seq] = Wt_v[v] . Xh_b[seq]     (V produced pre-transposed)
//           S[i][j]    = Q[i]    . K[j]          (causal tiles only)
//           O[i][v]    = P[i]    . Vt[v]         (K-loop causally truncated)
// Softmax: in place on S (f16 storage, f32 math), handles the causal mask.
//
// Every WMMA fragment load is a contiguous b128 pair; intermediates (~104 MB
// f16) are resident in the 192 MB L2, so the unfused pipeline pays ~no HBM tax.
// ---------------------------------------------------------------------------

typedef _Float16 half_t;
typedef __attribute__((ext_vector_type(8)))  float    f32x8;
typedef __attribute__((ext_vector_type(8)))  _Float16 f16x8;
typedef __attribute__((ext_vector_type(16))) _Float16 f16x16;

#define SEQ   2048
#define DIM   1024
#define BATCH 4

__device__ __forceinline__ f16x16 combine16(f16x8 lo, f16x8 hi) {
  return __builtin_shufflevector(lo, hi, 0,1,2,3,4,5,6,7,8,9,10,11,12,13,14,15);
}

// A-fragment (16x32 f16), per ISA layout: lane holds row lane%16;
// halves 0-7 = K[kk+hf*8 ..], halves 8-15 = K[kk+16+hf*8 ..], hf = lane/16.
__device__ __forceinline__ f16x16 load_a(const half_t* __restrict__ row, int kk, int hf) {
  f16x8 lo = *(const f16x8*)(row + kk + hf * 8);
  f16x8 hi = *(const f16x8*)(row + kk + 16 + hf * 8);
  return combine16(lo, hi);
}

__device__ __forceinline__ f32x8 wmma_f16(f16x16 a, f16x16 b, f32x8 c) {
  return __builtin_amdgcn_wmma_f32_16x16x32_f16(false, a, false, b, (short)0, c,
                                                false, false);
}

// C[32x64] += A[32xK] * B[64xK]^T ; A,B row-major f16 with given strides.
// B-fragment (32x16): lane holds column lane%16, same K layout as A-fragment
// (two b128 loads via load_a).
__device__ __forceinline__ void mma_tile(
    const half_t* __restrict__ A, size_t lda, int Mbase,
    const half_t* __restrict__ B, size_t ldb, int Nbase,
    int kbeg, int kend, int l16, int hf, f32x8 acc[2][4])
{
  const half_t* arow0 = A + (size_t)(Mbase + l16) * lda;
  const half_t* arow1 = A + (size_t)(Mbase + 16 + l16) * lda;
  const half_t* brow[4];
#pragma unroll
  for (int t = 0; t < 4; ++t)
    brow[t] = B + (size_t)(Nbase + t * 16 + l16) * ldb;

  for (int kk = kbeg; kk < kend; kk += 32) {
    f16x16 a0 = load_a(arow0, kk, hf);
    f16x16 a1 = load_a(arow1, kk, hf);
#pragma unroll
    for (int t = 0; t < 4; ++t) {
      f16x16 b = load_a(brow[t], kk, hf);
      acc[0][t] = wmma_f16(a0, b, acc[0][t]);
      acc[1][t] = wmma_f16(a1, b, acc[1][t]);
    }
  }
}

// ---------------------------------------------------------------------------
// Prep 1: x (f32) -> Xh (f16), flat elementwise, 8 elems/thread.
// ---------------------------------------------------------------------------
__global__ __launch_bounds__(256) void convert_x_kernel(
    const float* __restrict__ X, half_t* __restrict__ Xh)
{
  const size_t i = ((size_t)blockIdx.x * 256 + threadIdx.x) * 8;
  f32x8 v = *(const f32x8*)(X + i);
  *(f16x8*)(Xh + i) = __builtin_convertvector(v, f16x8);
}

// ---------------------------------------------------------------------------
// Prep 2: W[d][n] (f32) -> Wt[n][d] (f16), 32x32 LDS tile transpose.
// block = (32,8); grid = (DIM/32, DIM/32, 3).
// ---------------------------------------------------------------------------
__global__ __launch_bounds__(256) void transpose_w_kernel(
    const float* __restrict__ Wq, const float* __restrict__ Wk,
    const float* __restrict__ Wv, half_t* __restrict__ Wt)
{
  const int which = blockIdx.z;
  const float* W = (which == 0) ? Wq : ((which == 1) ? Wk : Wv);
  half_t* O = Wt + (size_t)which * DIM * DIM;

  const int nbase = blockIdx.x * 32;
  const int dbase = blockIdx.y * 32;
  __shared__ half_t tile[32][33];          // +1 pad: conflict-free transpose

#pragma unroll
  for (int r = 0; r < 4; ++r) {
    const int dl = threadIdx.y + r * 8;
    tile[dl][threadIdx.x] =
        (_Float16)W[(size_t)(dbase + dl) * DIM + nbase + threadIdx.x];
  }
  __syncthreads();
#pragma unroll
  for (int r = 0; r < 4; ++r) {
    const int nl = threadIdx.y + r * 8;
    O[(size_t)(nbase + nl) * DIM + dbase + threadIdx.x] =
        tile[threadIdx.x][nl];
  }
}

// ---------------------------------------------------------------------------
// GEMM A: Q and K projections. grid = (BATCH*SEQ/32, DIM/64, 2), block 32.
// ---------------------------------------------------------------------------
__global__ __launch_bounds__(32) void qk_proj_kernel(
    const half_t* __restrict__ Xh, const half_t* __restrict__ Wt,
    half_t* __restrict__ Q, half_t* __restrict__ Kc)
{
  const int lane = threadIdx.x, l16 = lane & 15, hf = lane >> 4;
  const int Mbase = blockIdx.x * 32;
  const int Nbase = blockIdx.y * 64;
  const int which = blockIdx.z;

  f32x8 acc[2][4] = {};
  mma_tile(Xh, DIM, Mbase, Wt + (size_t)which * DIM * DIM, DIM, Nbase,
           0, DIM, l16, hf, acc);

  half_t* O = (which == 0) ? Q : Kc;
#pragma unroll
  for (int s = 0; s < 2; ++s)
#pragma unroll
    for (int t = 0; t < 4; ++t)
#pragma unroll
      for (int r = 0; r < 8; ++r) {
        const int m = Mbase + s * 16 + hf * 8 + r;   // C/D: M = r + hf*8
        const int n = Nbase + t * 16 + l16;          // C/D: N = lane%16
        O[(size_t)m * DIM + n] = (_Float16)acc[s][t][r];
      }
}

// ---------------------------------------------------------------------------
// GEMM B: Vt[b][v][seq] = Wt_v[v] . Xh_b[seq]. grid = (DIM/32, SEQ/64, BATCH).
// ---------------------------------------------------------------------------
__global__ __launch_bounds__(32) void vt_proj_kernel(
    const half_t* __restrict__ Xh, const half_t* __restrict__ Wt,
    half_t* __restrict__ Vt)
{
  const int lane = threadIdx.x, l16 = lane & 15, hf = lane >> 4;
  const int Vbase = blockIdx.x * 32;
  const int Sbase = blockIdx.y * 64;
  const int b     = blockIdx.z;

  f32x8 acc[2][4] = {};
  mma_tile(Wt + (size_t)2 * DIM * DIM, DIM, Vbase,
           Xh + (size_t)b * SEQ * DIM, DIM, Sbase, 0, DIM, l16, hf, acc);

  half_t* O = Vt + (size_t)b * DIM * SEQ;
#pragma unroll
  for (int s = 0; s < 2; ++s)
#pragma unroll
    for (int t = 0; t < 4; ++t)
#pragma unroll
      for (int r = 0; r < 8; ++r) {
        const int v = Vbase + s * 16 + hf * 8 + r;
        const int q = Sbase + t * 16 + l16;
        O[(size_t)v * SEQ + q] = (_Float16)acc[s][t][r];
      }
}

// ---------------------------------------------------------------------------
// GEMM C: S = scale * Q K^T (f16 out). grid = (SEQ/32, SEQ/64, BATCH).
// ---------------------------------------------------------------------------
__global__ __launch_bounds__(32) void qk_scores_kernel(
    const half_t* __restrict__ Q, const half_t* __restrict__ Kc,
    half_t* __restrict__ S, const int* __restrict__ cm)
{
  const int lane = threadIdx.x, l16 = lane & 15, hf = lane >> 4;
  const int Ibase = blockIdx.x * 32;
  const int Jbase = blockIdx.y * 64;
  const int b     = blockIdx.z;
  if (cm[0] && Jbase > Ibase + 31) return;  // uniform: EXEC stays all-ones

  const half_t* Qb = Q  + (size_t)b * SEQ * DIM;
  const half_t* Kb = Kc + (size_t)b * SEQ * DIM;
  half_t*       Sb = S  + (size_t)b * SEQ * SEQ;

  f32x8 acc[2][4] = {};
  mma_tile(Qb, DIM, Ibase, Kb, DIM, Jbase, 0, DIM, l16, hf, acc);

  const float scale = 0.03125f;  // 1/sqrt(1024)
#pragma unroll
  for (int s = 0; s < 2; ++s)
#pragma unroll
    for (int t = 0; t < 4; ++t)
#pragma unroll
      for (int r = 0; r < 8; ++r) {
        const int i = Ibase + s * 16 + hf * 8 + r;
        const int j = Jbase + t * 16 + l16;
        Sb[(size_t)i * SEQ + j] = (_Float16)(acc[s][t][r] * scale);
      }
}

// ---------------------------------------------------------------------------
// Softmax rows in place; causal: read j<=i only, zero j>i.
// ---------------------------------------------------------------------------
__global__ __launch_bounds__(256) void softmax_rows_kernel(
    half_t* __restrict__ S, const int* __restrict__ cm)
{
  const int row = blockIdx.x;
  const int b = row >> 11, i = row & (SEQ - 1);
  half_t* Srow = S + (size_t)b * SEQ * SEQ + (size_t)i * SEQ;
  const int len = cm[0] ? (i + 1) : SEQ;
  const int tid = threadIdx.x;
  __shared__ float red[256];

  float m = -1e30f;
  for (int j = tid; j < len; j += 256) m = fmaxf(m, (float)Srow[j]);
  red[tid] = m; __syncthreads();
  for (int s = 128; s > 0; s >>= 1) {
    if (tid < s) red[tid] = fmaxf(red[tid], red[tid + s]);
    __syncthreads();
  }
  const float rowmax = red[0]; __syncthreads();

  float sum = 0.f;
  for (int j = tid; j < len; j += 256) sum += __expf((float)Srow[j] - rowmax);
  red[tid] = sum; __syncthreads();
  for (int s = 128; s > 0; s >>= 1) {
    if (tid < s) red[tid] += red[tid + s];
    __syncthreads();
  }
  const float inv = 1.0f / red[0];

  for (int j = tid; j < SEQ; j += 256) {
    float p = (j < len) ? __expf((float)Srow[j] - rowmax) * inv : 0.f;
    Srow[j] = (_Float16)p;
  }
}

// ---------------------------------------------------------------------------
// GEMM D: O = P V (f32 out). grid = (SEQ/32, DIM/64, BATCH).
// ---------------------------------------------------------------------------
__global__ __launch_bounds__(32) void pv_gemm_kernel(
    const half_t* __restrict__ P, const half_t* __restrict__ Vt,
    float* __restrict__ Out, const int* __restrict__ cm)
{
  const int lane = threadIdx.x, l16 = lane & 15, hf = lane >> 4;
  const int Ibase = blockIdx.x * 32;
  const int Vbase = blockIdx.y * 64;
  const int b     = blockIdx.z;

  const half_t* Pb  = P  + (size_t)b * SEQ * SEQ;
  const half_t* Vtb = Vt + (size_t)b * DIM * SEQ;
  const int kend = cm[0] ? (Ibase + 32) : SEQ;   // P[i][j>i] == 0

  f32x8 acc[2][4] = {};
  mma_tile(Pb, SEQ, Ibase, Vtb, SEQ, Vbase, 0, kend, l16, hf, acc);

  float* Ob = Out + (size_t)b * SEQ * DIM;
#pragma unroll
  for (int s = 0; s < 2; ++s)
#pragma unroll
    for (int t = 0; t < 4; ++t)
#pragma unroll
      for (int r = 0; r < 8; ++r) {
        const int i = Ibase + s * 16 + hf * 8 + r;
        const int v = Vbase + t * 16 + l16;
        Ob[(size_t)i * DIM + v] = acc[s][t][r];
      }
}

// ---------------------------------------------------------------------------
extern "C" void kernel_launch(void* const* d_in, const int* in_sizes, int n_in,
                              void* d_out, int out_size, void* d_ws, size_t ws_size,
                              hipStream_t stream) {
  const float* x  = (const float*)d_in[0];
  const float* Wq = (const float*)d_in[1];
  const float* Wk = (const float*)d_in[2];
  const float* Wv = (const float*)d_in[3];
  const int*   cm = (const int*)d_in[4];
  float* out = (float*)d_out;

  // Workspace (f16 elems): Xh | Wt(3) | Q | K | Vt | S   (~107 MB)
  half_t* ws = (half_t*)d_ws;
  const size_t XH_E  = (size_t)BATCH * SEQ * DIM;   // 8,388,608
  const size_t WT_E  = (size_t)3 * DIM * DIM;       // 3,145,728
  const size_t QKV_E = XH_E;
  half_t* Xh = ws;
  half_t* Wt = Xh + XH_E;
  half_t* Q  = Wt + WT_E;
  half_t* Kc = Q  + QKV_E;
  half_t* Vt = Kc + QKV_E;
  half_t* S  = Vt + QKV_E;                           // BATCH*SEQ*SEQ halves

  convert_x_kernel<<<dim3(XH_E / (256 * 8)), 256, 0, stream>>>(x, Xh);
  transpose_w_kernel<<<dim3(DIM / 32, DIM / 32, 3), dim3(32, 8), 0, stream>>>(
      Wq, Wk, Wv, Wt);
  qk_proj_kernel<<<dim3((BATCH * SEQ) / 32, DIM / 64, 2), 32, 0, stream>>>(
      Xh, Wt, Q, Kc);
  vt_proj_kernel<<<dim3(DIM / 32, SEQ / 64, BATCH), 32, 0, stream>>>(Xh, Wt, Vt);
  qk_scores_kernel<<<dim3(SEQ / 32, SEQ / 64, BATCH), 32, 0, stream>>>(
      Q, Kc, S, cm);
  softmax_rows_kernel<<<dim3(BATCH * SEQ), 256, 0, stream>>>(S, cm);
  pv_gemm_kernel<<<dim3(SEQ / 32, DIM / 64, BATCH), 32, 0, stream>>>(
      S, Vt, out, cm);
}
